// DFlashAttention_44375602102435
// MI455X (gfx1250) — matrix operand order repaired and checked by reading, not compiled, not run on hardware
//
#include <hip/hip_runtime.h>
#include <hip/hip_bf16.h>

typedef __attribute__((ext_vector_type(16))) __bf16 v16bf;
typedef __attribute__((ext_vector_type(8)))  float  v8f;

__device__ inline v8f wmma_bf16(v16bf a, v16bf b, v8f c) {
  return __builtin_amdgcn_wmma_f32_16x16x32_bf16(false, a, false, b, (short)0, c, false, false);
}

union Frag {
  v16bf v;
  uint4 u[2];
  __device__ Frag() {}
};

// 16B/lane (512B/wave) global->LDS async copy. LDS operand: low 32 bits of the
// flat pointer are the wave-relative LDS byte offset (ISA 10.2 aperture rules).
__device__ inline void async_ld16(void* lds, const void* g) {
  asm volatile("global_load_async_to_lds_b128 %0, %1, off"
               :: "v"((unsigned)(uintptr_t)lds), "v"((unsigned long long)(uintptr_t)g)
               : "memory");
}
__device__ inline void wait_async_le6() { asm volatile("s_wait_asynccnt 0x6" ::: "memory"); }
__device__ inline void wait_async_le4() { asm volatile("s_wait_asynccnt 0x4" ::: "memory"); }
__device__ inline void wait_async_0()   { asm volatile("s_wait_asynccnt 0x0" ::: "memory"); }

// ---------------------------------------------------------------- f32 -> bf16
__global__ __launch_bounds__(256) void cvt_f32_bf16(const float* __restrict__ in,
                                                    __bf16* __restrict__ out, int n4) {
  int i = blockIdx.x * blockDim.x + threadIdx.x;
  if (i >= n4) return;
  const float4 f = ((const float4*)in)[i];
  __bf16 o[4] = {(__bf16)f.x, (__bf16)f.y, (__bf16)f.z, (__bf16)f.w};
  ((uint2*)out)[i] = *(const uint2*)o;
}

// ------------------------------------------------ TN GEMM: C[M,N]=A[M,K]*B[N,K]^T
// block: 256 thr = 8 waves; tile 256(M) x 128(N), K-step 32, async double-buffer;
// wave: 64x64 (4x4 wmma per K-step).
__global__ __launch_bounds__(256) void gemm_bf16_tn(const __bf16* __restrict__ A,
                                                    const __bf16* __restrict__ B,
                                                    float* __restrict__ C,
                                                    int M, int N, int K) {
  __shared__ __bf16 As[2][256][32];   // 32 KB
  __shared__ __bf16 Bs[2][128][32];   // 16 KB
  const int tid  = threadIdx.x;
  const int lane = tid & 31;
  const int wave = tid >> 5;
  const int wm = wave & 3;            // 4 waves down M (64 rows each)
  const int wn = wave >> 2;           // 2 waves across N (64 cols each)
  const int fr = lane & 15;
  const int hi = lane >> 4;
  const int bm = blockIdx.y * 256;
  const int bn = blockIdx.x * 128;

  // async staging coords: 4 lanes per 64B row-chunk
  const int srow = lane >> 2;         // 0..7
  const int scol = (lane & 3) * 8;    // element col (16B granules)

  auto stage = [&](int k0, int p) {
    #pragma unroll
    for (int i = 0; i < 4; ++i) {     // A: wave covers 32 rows
      const int row = wave * 32 + i * 8 + srow;
      async_ld16(&As[p][row][scol], A + (size_t)(bm + row) * K + k0 + scol);
    }
    #pragma unroll
    for (int i = 0; i < 2; ++i) {     // B: wave covers 16 rows
      const int row = wave * 16 + i * 8 + srow;
      async_ld16(&Bs[p][row][scol], B + (size_t)(bn + row) * K + k0 + scol);
    }
  };

  v8f acc[4][4];
  #pragma unroll
  for (int mi = 0; mi < 4; ++mi)
    #pragma unroll
    for (int ni = 0; ni < 4; ++ni)
      #pragma unroll
      for (int r = 0; r < 8; ++r) acc[mi][ni][r] = 0.0f;

  stage(0, 0);
  const int nk = K >> 5;
  for (int t = 0; t < nk; ++t) {
    const int p = t & 1;
    if (t + 1 < nk) { stage((t + 1) << 5, p ^ 1); wait_async_le6(); }
    else            { wait_async_0(); }
    __syncthreads();

    Frag fa[4], fb[4];
    #pragma unroll
    for (int mi = 0; mi < 4; ++mi) {
      const __bf16* pA = &As[p][wm * 64 + mi * 16 + fr][0];
      fa[mi].u[0] = *(const uint4*)(pA + hi * 8);
      fa[mi].u[1] = *(const uint4*)(pA + 16 + hi * 8);
    }
    #pragma unroll
    for (int ni = 0; ni < 4; ++ni) {
      const __bf16* pB = &Bs[p][wn * 64 + ni * 16 + fr][0];
      fb[ni].u[0] = *(const uint4*)(pB + hi * 8);
      fb[ni].u[1] = *(const uint4*)(pB + 16 + hi * 8);
    }
    #pragma unroll
    for (int mi = 0; mi < 4; ++mi)
      #pragma unroll
      for (int ni = 0; ni < 4; ++ni)
        acc[mi][ni] = wmma_bf16(fa[mi].v, fb[ni].v, acc[mi][ni]);
    __syncthreads();
  }

  #pragma unroll
  for (int mi = 0; mi < 4; ++mi)
    #pragma unroll
    for (int ni = 0; ni < 4; ++ni) {
      const int col = bn + wn * 64 + ni * 16 + fr;
      #pragma unroll
      for (int r = 0; r < 8; ++r) {
        const int row = bm + wm * 64 + mi * 16 + r + hi * 8;
        C[(size_t)row * N + col] = acc[mi][ni][r];
      }
    }
}

// -------------------------------- RMSNorm + RoPE + layout (Q,K) / transpose (V)
__global__ __launch_bounds__(64) void qkv_prep(const float* __restrict__ Qf,
                                               const float* __restrict__ Kf,
                                               const float* __restrict__ Vf,
                                               const int* __restrict__ positions,
                                               const float* __restrict__ qw,
                                               const float* __restrict__ kw,
                                               __bf16* __restrict__ Qb,   // [B,32,S,128]
                                               __bf16* __restrict__ Kb,   // [B,8,S,128]
                                               __bf16* __restrict__ Vt) { // [B,8,128,S]
  const int token = blockIdx.x;
  const int h = blockIdx.y;            // 0..47
  const int b = token >> 10, s = token & 1023;
  const int d = threadIdx.x;           // 0..63

  if (h >= 40) {                       // V: convert + transpose
    const int kvh = h - 40;
    const float* src = Vf + (size_t)token * 1024 + kvh * 128;
    __bf16* dst = Vt + ((size_t)(b * 8 + kvh) * 128) * 1024 + s;
    dst[(size_t)d * 1024]        = (__bf16)src[d];
    dst[(size_t)(d + 64) * 1024] = (__bf16)src[d + 64];
    return;
  }
  const bool isq = h < 32;
  const float* src = isq ? Qf + (size_t)token * 4096 + h * 128
                         : Kf + (size_t)token * 1024 + (h - 32) * 128;
  const float* w = isq ? qw : kw;

  float x1 = src[d], x2 = src[d + 64];
  float ss = x1 * x1 + x2 * x2;
  #pragma unroll
  for (int m = 16; m >= 1; m >>= 1) ss += __shfl_xor(ss, m);
  __shared__ float red[2];
  if ((threadIdx.x & 31) == 0) red[threadIdx.x >> 5] = ss;
  __syncthreads();
  const float inv = rsqrtf((red[0] + red[1]) * (1.0f / 128.0f) + 1e-6f);
  x1 = x1 * inv * w[d];
  x2 = x2 * inv * w[d + 64];

  const float pos  = (float)positions[s];
  const float freq = exp2f(-((float)d * (1.0f / 64.0f)) * 19.931568569324174f);
  float sn, cs;
  __sincosf(pos * freq, &sn, &cs);
  float o1 = x1 * cs - x2 * sn;
  float o2 = x2 * cs + x1 * sn;

  if (isq) {
    o1 *= 0.08838834764831845f;        // headdim^-0.5 folded into Q
    o2 *= 0.08838834764831845f;
    __bf16* dst = Qb + ((size_t)(b * 32 + h) * 1024 + s) * 128;
    dst[d] = (__bf16)o1; dst[d + 64] = (__bf16)o2;
  } else {
    __bf16* dst = Kb + ((size_t)(b * 8 + (h - 32)) * 1024 + s) * 128;
    dst[d] = (__bf16)o1; dst[d + 64] = (__bf16)o2;
  }
}

// ------------------------------------------- flash attention, wave = 16 queries
// grid (S/128, H, B); block 256 (8 waves). K/V staged in LDS via async copies
// (double-buffered); online softmax; P transposed through per-wave LDS tile.
__global__ __launch_bounds__(256) void flash_attn(const __bf16* __restrict__ Qb,
                                                  const __bf16* __restrict__ Kb,
                                                  const __bf16* __restrict__ Vt,
                                                  __bf16* __restrict__ O) { // [B,S,H,D]
  constexpr int S = 1024, D = 128, H = 32, KVH = 8;
  const int lane = threadIdx.x & 31;
  const int wave = threadIdx.x >> 5;
  const int fr = lane & 15, hi = lane >> 4;
  const int b = blockIdx.z, h = blockIdx.y;
  const int q0 = blockIdx.x * 128 + wave * 16;
  const int kvh = h >> 2;

  const __bf16* Qp = Qb + ((size_t)(b * H + h) * S + q0) * D;
  const __bf16* Kp = Kb + ((size_t)(b * KVH + kvh) * S) * D;
  const __bf16* Vp = Vt + ((size_t)(b * KVH + kvh) * D) * S;

  __shared__ __bf16 Ks[2][32][128];   // 16 KB: 32 keys x 128 d
  __shared__ __bf16 Vs[2][128][32];   // 16 KB: 128 d x 32 keys
  __shared__ __bf16 Pl[8][16][32];    //  8 KB: per-wave P staging

  auto stage = [&](int kb, int p) {
    #pragma unroll
    for (int i = 0; i < 2; ++i) {     // K: wave covers 4 key rows (256B each)
      const int key = wave * 4 + i * 2 + (lane >> 4);
      const int col = (lane & 15) * 8;
      async_ld16(&Ks[p][key][col], Kp + (size_t)(kb + key) * D + col);
    }
    #pragma unroll
    for (int i = 0; i < 2; ++i) {     // V: wave covers 16 d rows (64B each)
      const int row = wave * 16 + i * 8 + (lane >> 2);
      const int col = (lane & 3) * 8;
      async_ld16(&Vs[p][row][col], Vp + (size_t)row * S + kb + col);
    }
  };

  Frag aq[4];
  #pragma unroll
  for (int kk = 0; kk < 4; ++kk) {
    const __bf16* p = Qp + (size_t)fr * D + kk * 32;
    aq[kk].u[0] = *(const uint4*)(p + hi * 16);
    aq[kk].u[1] = *(const uint4*)(p + hi * 16 + 8);
  }

  v8f acc[8];
  float mrow[8], lrow[8];
  #pragma unroll
  for (int j = 0; j < 8; ++j)
    #pragma unroll
    for (int r = 0; r < 8; ++r) acc[j][r] = 0.0f;
  #pragma unroll
  for (int r = 0; r < 8; ++r) { mrow[r] = -1e30f; lrow[r] = 0.0f; }

  stage(0, 0);
  constexpr int NT = S / 32;
  for (int t = 0; t < NT; ++t) {
    const int p = t & 1;
    if (t + 1 < NT) { stage((t + 1) * 32, p ^ 1); wait_async_le4(); }
    else            { wait_async_0(); }
    __syncthreads();

    v8f sc0, sc1;
    #pragma unroll
    for (int r = 0; r < 8; ++r) { sc0[r] = 0.0f; sc1[r] = 0.0f; }
    #pragma unroll
    for (int kk = 0; kk < 4; ++kk) {
      Frag bk0, bk1;
      const __bf16* p0 = &Ks[p][fr][kk * 32 + hi * 16];
      const __bf16* p1 = &Ks[p][16 + fr][kk * 32 + hi * 16];
      bk0.u[0] = *(const uint4*)p0; bk0.u[1] = *(const uint4*)(p0 + 8);
      bk1.u[0] = *(const uint4*)p1; bk1.u[1] = *(const uint4*)(p1 + 8);
      sc0 = wmma_bf16(aq[kk].v, bk0.v, sc0);
      sc1 = wmma_bf16(aq[kk].v, bk1.v, sc1);
    }
    // online softmax: row m = r + 8*hi lives within a 16-lane half
    #pragma unroll
    for (int r = 0; r < 8; ++r) {
      float v0 = sc0[r], v1 = sc1[r];
      float mx = fmaxf(v0, v1);
      mx = fmaxf(mx, __shfl_xor(mx, 1));
      mx = fmaxf(mx, __shfl_xor(mx, 2));
      mx = fmaxf(mx, __shfl_xor(mx, 4));
      mx = fmaxf(mx, __shfl_xor(mx, 8));
      const float mnew  = fmaxf(mrow[r], mx);
      const float scale = __expf(mrow[r] - mnew);
      mrow[r] = mnew;
      const float e0 = __expf(v0 - mnew);
      const float e1 = __expf(v1 - mnew);
      float rs = e0 + e1;
      rs += __shfl_xor(rs, 1);
      rs += __shfl_xor(rs, 2);
      rs += __shfl_xor(rs, 4);
      rs += __shfl_xor(rs, 8);
      lrow[r] = lrow[r] * scale + rs;
      #pragma unroll
      for (int j = 0; j < 8; ++j) acc[j][r] *= scale;
      const int m = r + hi * 8;
      Pl[wave][m][fr]      = (__bf16)e0;
      Pl[wave][m][fr + 16] = (__bf16)e1;
    }
    asm volatile("s_wait_dscnt 0x0" ::: "memory");   // wave-private LDS RAW
    Frag pf;
    {
      const __bf16* pp = &Pl[wave][fr][0];
      pf.u[0] = *(const uint4*)(pp + hi * 16);
      pf.u[1] = *(const uint4*)(pp + hi * 16 + 8);
    }
    #pragma unroll
    for (int j = 0; j < 8; ++j) {      // O[16q,128d] += P[16,32] @ V[32,16d]
      Frag bv;
      const __bf16* pv = &Vs[p][j * 16 + fr][hi * 16];
      bv.u[0] = *(const uint4*)pv; bv.u[1] = *(const uint4*)(pv + 8);
      acc[j] = wmma_bf16(pf.v, bv.v, acc[j]);
    }
    __syncthreads();
  }

  #pragma unroll
  for (int r = 0; r < 8; ++r) {
    const float linv = 1.0f / lrow[r];
    const int m = r + hi * 8;
    const size_t base = (((size_t)(b * S + q0 + m)) * H + h) * D;
    #pragma unroll
    for (int j = 0; j < 8; ++j)
      O[base + j * 16 + fr] = (__bf16)(acc[j][r] * linv);
  }
}

// --------------------------------------------------------------------- launch
extern "C" void kernel_launch(void* const* d_in, const int* in_sizes, int n_in,
                              void* d_out, int out_size, void* d_ws, size_t ws_size,
                              hipStream_t stream) {
  constexpr int B = 4, S = 1024, HID = 4096, NH = 32, NKV = 8, HD = 128;
  constexpr int T = B * S;
  const int*   positions = (const int*)d_in[0];
  const float* hs = (const float*)d_in[1];
  const float* wq = (const float*)d_in[2];
  const float* wk = (const float*)d_in[3];
  const float* wv = (const float*)d_in[4];
  const float* wo = (const float*)d_in[5];
  const float* qw = (const float*)d_in[6];
  const float* kw = (const float*)d_in[7];
  float* out = (float*)d_out;

  char* ws = (char*)d_ws;
  size_t off = 0;
  auto alloc = [&](size_t bytes) { void* p = ws + off; off += (bytes + 255) & ~(size_t)255; return p; };
  __bf16* hsb  = (__bf16*)alloc((size_t)T * HID * 2);        // reused later as Qb
  __bf16* W    = (__bf16*)alloc((size_t)HID * HID * 2);      // current weight (bf16)
  float*  Qf   = (float*) alloc((size_t)T * HID * 4);        // reused later as attn_out
  float*  Kf   = (float*) alloc((size_t)T * NKV * HD * 4);
  float*  Vf   = (float*) alloc((size_t)T * NKV * HD * 4);
  __bf16* Kb   = (__bf16*)alloc((size_t)B * NKV * S * HD * 2);
  __bf16* Vt   = (__bf16*)alloc((size_t)B * NKV * S * HD * 2);
  __bf16* Qb   = hsb;
  __bf16* attn = (__bf16*)Qf;

  auto cvt = [&](const float* src, __bf16* dst, int n) {
    int n4 = n / 4;
    cvt_f32_bf16<<<(n4 + 255) / 256, 256, 0, stream>>>(src, dst, n4);
  };

  cvt(hs, hsb, T * HID);

  cvt(wq, W, NH * HD * HID);
  gemm_bf16_tn<<<dim3(HID / 128, T / 256), 256, 0, stream>>>(hsb, W, Qf, T, HID, HID);
  cvt(wk, W, NKV * HD * HID);
  gemm_bf16_tn<<<dim3((NKV * HD) / 128, T / 256), 256, 0, stream>>>(hsb, W, Kf, T, NKV * HD, HID);
  cvt(wv, W, NKV * HD * HID);
  gemm_bf16_tn<<<dim3((NKV * HD) / 128, T / 256), 256, 0, stream>>>(hsb, W, Vf, T, NKV * HD, HID);

  qkv_prep<<<dim3(T, 48), 64, 0, stream>>>(Qf, Kf, Vf, positions, qw, kw, Qb, Kb, Vt);

  cvt(wo, W, HID * NH * HD);

  flash_attn<<<dim3(S / 128, NH, B), 256, 0, stream>>>(Qb, Kb, Vt, attn);

  gemm_bf16_tn<<<dim3(HID / 128, T / 256), 256, 0, stream>>>(attn, W, out, T, HID, HID);
}
